// ResidualSieveKAN_21526376087735
// MI455X (gfx1250) — matrix operand order repaired and checked by reading, not compiled, not run on hardware
//
#include <hip/hip_runtime.h>
#include <hip/hip_bf16.h>
#include <stdint.h>

// CDNA5 / gfx1250 KAN forward: fused basis-expansion + bf16 WMMA GEMM with
// async global->LDS double-buffered weight staging (ASYNCcnt path) and
// register-level fragment pipelining (loads for chunk c+1 shadow the 4
// back-to-back v_wmma of chunk c).

typedef __bf16 bf16_t;
typedef bf16_t v16bf __attribute__((ext_vector_type(16)));
typedef float  v8f   __attribute__((ext_vector_type(8)));

#define TILE_M     64
#define TILE_N     64
#define SEG        128     // padded per-feature K segment
#define NBASIS     103     // valid basis functions (G + K = 100 + 3)
#define SILU_SLOT  103     // base-path column within the segment
#define LDS_STRIDE 136     // SEG + 8: 272B row stride, conflict-free b128
#define TILE_ELEMS (TILE_N * SEG)        // 8192 bf16 per (feature, ntile)
#define TILE_BYTES (TILE_ELEMS * 2)      // 16 KB

union FragU { v16bf v; uint4 q[2]; };

// One async b128 copy: LDS[lds + imm] = MEM[saddr + voff + imm], 16 bytes/lane.
#define ASYNC_CP16(LDS, VOFF, BASE, IMM)                                   \
    asm volatile("global_load_async_to_lds_b128 %0, %1, %2 offset:" #IMM  \
                 :: "v"(LDS), "v"(VOFF), "s"(BASE) : "memory")

// Copy 128 contiguous bytes (half of one 128-elem weight row) per thread.
__device__ __forceinline__
void async_copy_half_row(uint32_t lds, uint32_t voff, const void* base)
{
    ASYNC_CP16(lds, voff, base, 0);
    ASYNC_CP16(lds, voff, base, 16);
    ASYNC_CP16(lds, voff, base, 32);
    ASYNC_CP16(lds, voff, base, 48);
    ASYNC_CP16(lds, voff, base, 64);
    ASYNC_CP16(lds, voff, base, 80);
    ASYNC_CP16(lds, voff, base, 96);
    ASYNC_CP16(lds, voff, base, 112);
}

// ---------------------------------------------------------------------------
// Weight pre-pack: fp32 (sw, bw) -> bf16, layout [ntile][i][n:64][k:128] so a
// (feature, ntile) tile is one contiguous 16 KB block for the async engine.
// ---------------------------------------------------------------------------
__global__ __launch_bounds__(256)
void pack_weights_kernel(const float* __restrict__ sw,
                         const float* __restrict__ bw,
                         bf16_t* __restrict__ wp,
                         int fin, int fout, long total)
{
    long idx = (long)blockIdx.x * 256 + threadIdx.x;
    if (idx >= total) return;
    const int  k    = (int)(idx & (SEG - 1));
    const int  n    = (int)((idx >> 7) & (TILE_N - 1));
    const long rest = idx >> 13;
    const int  i    = (int)(rest % fin);
    const int  nt   = (int)(rest / fin);
    const int  ng   = nt * TILE_N + n;
    float v = 0.0f;
    if (ng < fout) {
        if (k < NBASIS)          v = sw[((size_t)ng * fin + i) * NBASIS + k];
        else if (k == SILU_SLOT) v = bw[(size_t)ng * fin + i];
    }
    wp[idx] = (bf16_t)v;
}

// ---------------------------------------------------------------------------
// Fused KAN layer.
// ---------------------------------------------------------------------------
template<int F_IN, int F_OUT, bool RES>
__global__ __launch_bounds__(128)
void kan_layer_kernel(const float* __restrict__ hin,
                      const bf16_t* __restrict__ wp,
                      const float* __restrict__ wbp,
                      const float* __restrict__ wsp,
                      float* __restrict__ hout)
{
    __shared__ bf16_t As[TILE_M * LDS_STRIDE];       // 17.4 KB basis tile
    __shared__ bf16_t Bs[2][TILE_N * LDS_STRIDE];    // 2 x 17.4 KB weight tiles

    const int tid    = threadIdx.x;
    const int lane   = tid & 31;
    const int wave   = tid >> 5;                 // 4 waves, 16 rows each
    const int blockM = blockIdx.x * TILE_M;      // batch rows
    const int blockN = blockIdx.y * TILE_N;      // output cols

    const float wbv = wbp[0];
    const float wsv = wsp[0];

    v8f acc[4];
    #pragma unroll
    for (int t = 0; t < 4; ++t) {
        #pragma unroll
        for (int e = 0; e < 8; ++e) acc[t][e] = 0.0f;
    }

    // Fragment addressing (CDNA5 16-bit WMMA VGPR layouts):
    // A 16x32: lanes 0-15 = M rows, K halves {0..7,16..23}; lanes 16-31 same
    //          rows, K halves {8..15,24..31} -> half split at +8 elements.
    // B 32x16: lanes 0-15 = N cols, K {0..7,16..23}; lanes 16-31 same
    //          cols, K {8..15,24..31} -> half split at +8 elements.
    const int aRow = wave * 16 + (lane & 15);
    const int aKlo = (lane >> 4) * 8;
    const int bKlo = (lane >> 4) * 8;
    const int bCol = lane & 15;

    // Async staging assignment: 2 threads per weight row, 64 elems (128B) each
    const int wn   = tid >> 1;
    const int whal = tid & 1;
    const bf16_t* wtile = wp + (size_t)blockIdx.y * F_IN * TILE_ELEMS;
    uint32_t voff = (uint32_t)((wn * SEG + whal * 64) * 2);  // bytes, feature 0
    const uint32_t ldsElem = (uint32_t)(wn * LDS_STRIDE + whal * 64);
    const uint32_t ldsB[2] = {
        (uint32_t)(uintptr_t)&Bs[0][ldsElem],    // low 32 bits = LDS offset
        (uint32_t)(uintptr_t)&Bs[1][ldsElem]
    };

    // Prologue: stream feature 0's weight tile into buffer 0.
    async_copy_half_row(ldsB[0], voff, (const void*)wtile);
    voff += TILE_BYTES;

    for (int i = 0; i < F_IN; ++i) {
        // kick off next feature's weight tile (double buffer)
        const bool more = (i + 1 < F_IN);
        if (more) {
            async_copy_half_row(ldsB[(i + 1) & 1], voff, (const void*)wtile);
            voff += TILE_BYTES;
        }

        // ---- stage A tile: sparse cubic B-spline basis (overlaps async) ----
        if (tid < TILE_M) {
            const int   r = tid;
            const float x = hin[(size_t)(blockM + r) * F_IN + i];
            uint4* prow = (uint4*)&As[r * LDS_STRIDE];
            #pragma unroll
            for (int q = 0; q < (SEG * 2) / 16; ++q) prow[q] = make_uint4(0,0,0,0);
            const float sig = 1.0f / (1.0f + __expf(-x));
            As[r * LDS_STRIDE + SILU_SLOT] = (bf16_t)(wbv * x * sig);
            const float xs = (x + 0.5f) * 50.0f;          // 1/H = G/2 = 50
            if (xs >= -3.0f && xs < 103.0f) {
                const float cf   = floorf(xs);
                const int   cell = (int)cf;
                const float u    = xs - cf;
                const float um   = 1.0f - u;
                const float u2   = u * u, u3 = u2 * u;
                float bvs[4];
                bvs[0] = um * um * um * (1.0f / 6.0f);
                bvs[1] = (3.0f * u3 - 6.0f * u2 + 4.0f) * (1.0f / 6.0f);
                bvs[2] = (-3.0f * u3 + 3.0f * u2 + 3.0f * u + 1.0f) * (1.0f / 6.0f);
                bvs[3] = u3 * (1.0f / 6.0f);
                #pragma unroll
                for (int jj = 0; jj < 4; ++jj) {
                    const int j = cell + jj;
                    if (j >= 0 && j < NBASIS)
                        As[r * LDS_STRIDE + j] = (bf16_t)(wsv * bvs[jj]);
                }
            }
        }

        // wait for feature i's weight tile (8 async issues may remain in
        // flight for feature i+1); completion is in-order per wave.
        if (more) asm volatile("s_wait_asynccnt 0x8" ::: "memory");
        else      asm volatile("s_wait_asynccnt 0x0" ::: "memory");
        __syncthreads();

        // ---- WMMA: 4 K-chunks of 32, fragment loads pipelined one chunk
        //      ahead so each load group is shadowed by 4 wmmas --------------
        const bf16_t* bbuf = &Bs[i & 1][0];
        FragU aF[2], bF[2][4];

        // load chunk 0 fragments
        {
            const uint4* pa = (const uint4*)&As[aRow * LDS_STRIDE + aKlo];
            aF[0].q[0] = pa[0];
            aF[0].q[1] = pa[2];
            #pragma unroll
            for (int ct = 0; ct < 4; ++ct) {
                const uint4* pb =
                    (const uint4*)&bbuf[(ct * 16 + bCol) * LDS_STRIDE + bKlo];
                bF[0][ct].q[0] = pb[0];
                bF[0][ct].q[1] = pb[2];
            }
        }

        #pragma unroll
        for (int c = 0; c < 4; ++c) {
            const int cur = c & 1;
            const int nxt = cur ^ 1;
            if (c < 3) {   // prefetch chunk c+1 fragments into the other buffer
                const uint4* pa =
                    (const uint4*)&As[aRow * LDS_STRIDE + (c + 1) * 32 + aKlo];
                aF[nxt].q[0] = pa[0];
                aF[nxt].q[1] = pa[2];
                #pragma unroll
                for (int ct = 0; ct < 4; ++ct) {
                    const uint4* pb = (const uint4*)
                        &bbuf[(ct * 16 + bCol) * LDS_STRIDE + (c + 1) * 32 + bKlo];
                    bF[nxt][ct].q[0] = pb[0];
                    bF[nxt][ct].q[1] = pb[2];
                }
            }
            #pragma unroll
            for (int ct = 0; ct < 4; ++ct) {
                acc[ct] = __builtin_amdgcn_wmma_f32_16x16x32_bf16(
                    false, aF[cur].v, false, bF[cur][ct].v, (short)0,
                    acc[ct], false, false);
            }
        }
        __syncthreads();
    }

    // ---- epilogue: C/D layout -> global, optional residual -----------------
    const int mSub = (lane >> 4) * 8;     // lanes 16-31 hold M rows 8..15
    #pragma unroll
    for (int ct = 0; ct < 4; ++ct) {
        const int ng = blockN + ct * 16 + (lane & 15);
        if (ng < F_OUT) {
            #pragma unroll
            for (int v = 0; v < 8; ++v) {
                const int b = blockM + wave * 16 + mSub + v;
                float val = acc[ct][v];
                if (RES) val += hin[(size_t)b * F_IN + ng];
                hout[(size_t)b * F_OUT + ng] = val;
            }
        }
    }
}

extern "C" void kernel_launch(void* const* d_in, const int* in_sizes, int n_in,
                              void* d_out, int out_size, void* d_ws, size_t ws_size,
                              hipStream_t stream)
{
    const float* x = (const float*)d_in[0];
    const float *BW[6], *SW[6], *WB[6], *WS[6];
    for (int k = 0; k < 6; ++k) {
        BW[k] = (const float*)d_in[1 + 4 * k];
        SW[k] = (const float*)d_in[2 + 4 * k];
        WB[k] = (const float*)d_in[3 + 4 * k];
        WS[k] = (const float*)d_in[4 + 4 * k];
    }
    // workspace: h0,h1 fp32 ping-pong (2 x ~2MB) + packed bf16 weights (~16MB)
    float*  h0 = (float*)d_ws;
    float*  h1 = h0 + (size_t)2048 * 243;
    bf16_t* wpk = (bf16_t*)(h1 + (size_t)2048 * 243);

    dim3 blk128(128), blk256(256);
    dim3 gridMid(2048 / TILE_M, 4);      // 4 N-tiles cover 256 >= 243 cols
    dim3 gridLast(2048 / TILE_M, 1);

    const int fins[6]  = { 64, 243, 243, 243, 243, 243 };
    const int fouts[6] = { 243, 243, 243, 243, 243, 1 };
    const int nts[6]   = { 4, 4, 4, 4, 4, 1 };

    const float* hsrc[6] = { x, h0, h1, h0, h1, h0 };
    float*       hdst[6] = { h0, h1, h0, h1, h0, (float*)d_out };

    for (int k = 0; k < 6; ++k) {
        const long total = (long)nts[k] * fins[k] * TILE_ELEMS;
        pack_weights_kernel<<<dim3((unsigned)((total + 255) / 256)), blk256, 0, stream>>>(
            SW[k], BW[k], wpk, fins[k], fouts[k], total);
        dim3 grid = (k == 5) ? gridLast : gridMid;
        switch (k) {
        case 0:
            kan_layer_kernel< 64, 243, false><<<grid, blk128, 0, stream>>>(
                hsrc[k], wpk, WB[k], WS[k], hdst[k]);
            break;
        case 5:
            kan_layer_kernel<243,   1, false><<<grid, blk128, 0, stream>>>(
                hsrc[k], wpk, WB[k], WS[k], hdst[k]);
            break;
        default:
            kan_layer_kernel<243, 243, true ><<<grid, blk128, 0, stream>>>(
                hsrc[k], wpk, WB[k], WS[k], hdst[k]);
            break;
        }
    }
}
